// LSTM_41583873359969
// MI455X (gfx1250) — matrix operand order repaired and checked by reading, not compiled, not run on hardware
//
#include <hip/hip_runtime.h>

// Problem constants (from reference): T=1024, B=64, H=512, L=2
#define T_STEPS 1024
#define B_SZ    64
#define H_SZ    512
#define L_SZ    2
#define G4      (4 * H_SZ)        // 2048 gate columns
#define KSTEP   32                // K per bf16 WMMA
#define AS_STRIDE (H_SZ + 8)      // padded LDS row stride (bf16 elems) -> conflict-free b128 reads
#define SCAN_BLOCKS 16            // persistent scan grid (must all be co-resident)

typedef __attribute__((ext_vector_type(16))) __bf16 v16bf;
typedef __attribute__((ext_vector_type(8)))  float  v8f;

union FragBF { v16bf v; uint4 q[2]; };

__device__ __forceinline__ float sigmoidf_(float x) {
    return 1.0f / (1.0f + __expf(-x));
}

// ---------------- small prep kernels ----------------

__global__ void k_f32_to_bf16(const float* __restrict__ src, __bf16* __restrict__ dst, int n) {
    int i = blockIdx.x * blockDim.x + threadIdx.x;
    if (i < n) dst[i] = (__bf16)src[i];
}

// W: [G4, H] row-major fp32  ->  Wt: [H, G4] row-major bf16 (K-major for B fragments)
__global__ void k_transpose_convert(const float* __restrict__ W, __bf16* __restrict__ Wt) {
    int idx = blockIdx.x * blockDim.x + threadIdx.x;   // over H*G4, Wt-linear (coalesced writes)
    int g = idx % G4;
    int k = idx / G4;
    Wt[idx] = (__bf16)W[(size_t)g * H_SZ + k];
}

__global__ void k_bias_sum(const float* __restrict__ bi, const float* __restrict__ bh,
                           float* __restrict__ bs) {
    int i = blockIdx.x * blockDim.x + threadIdx.x;
    if (i < G4) bs[i] = bi[i] + bh[i];
}

__global__ void k_zero_u32(unsigned int* p) {
    if (threadIdx.x == 0 && blockIdx.x == 0) *p = 0u;
}

// ---------------- WMMA fragment helpers ----------------

// A fragment (16x32 bf16) from LDS tile, ISA 7.12.2 layout:
// lanes 0-15: row=lane, K = k0..k0+7 (q0), k0+16..k0+23 (q1)
// lanes 16-31: row=lane-16, K = k0+8..k0+15 (q0), k0+24..k0+31 (q1)
__device__ __forceinline__ v16bf frag_a(const __bf16* As, int lane, int k0) {
    int row = lane & 15, half = lane >> 4;
    FragBF f;
    f.q[0] = *(const uint4*)(As + row * AS_STRIDE + k0 + half * 8);
    f.q[1] = *(const uint4*)(As + row * AS_STRIDE + k0 + 16 + half * 8);
    return f.v;
}

// B fragment (32x16 bf16) from K-major Wt[H][G4]: col = lane&15, K in same order as frag_a.
__device__ __forceinline__ v16bf frag_b(const __bf16* __restrict__ Wt, int lane, int k0, int col0) {
    const __bf16* p = Wt + (size_t)k0 * G4 + col0 + (lane & 15);
    int half = lane >> 4;
    FragBF f;
#pragma unroll
    for (int s = 0; s < 16; ++s) {
        int k = (s < 8) ? (8 * half + s) : (16 + 8 * half + (s - 8));
        f.v[s] = p[(size_t)k * G4];
    }
    return f.v;
}

// Stage a 16xH bf16 A-tile (row stride H) into padded LDS; 256 threads, 16B chunks.
__device__ __forceinline__ void stage_a(const __bf16* __restrict__ Ag, __bf16* As) {
    for (int c = threadIdx.x; c < 16 * 64; c += 256) {   // 64 x 16B chunks per row
        int row = c >> 6;
        int off = (c & 63) * 8;                          // bf16 elems
        *(uint4*)(As + row * AS_STRIDE + off) = *(const uint4*)(Ag + (size_t)row * H_SZ + off);
    }
}

// ---------------- Phase 1: xg[m, 4H] = Xbf[m, :] @ Wt + (b_ih + b_hh) ----------------
// grid: (M/16)*4 blocks, 256 threads (8 waves). Wave computes one 16x16 tile per gate (4 accs).
__global__ __launch_bounds__(256) void k_gates_pre(
    const __bf16* __restrict__ Xbf,   // [M, H] bf16
    const __bf16* __restrict__ Wt,    // [H, G4] bf16
    const float*  __restrict__ bias,  // [G4]
    float*        __restrict__ xg)    // [M, G4]
{
    __shared__ __align__(16) __bf16 As[16 * AS_STRIDE];
    int wave = threadIdx.x >> 5, lane = threadIdx.x & 31;
    int mt = blockIdx.x >> 2, ntg = blockIdx.x & 3;
    int nt = ntg * 8 + wave;                 // 0..31 (columns within each gate group)
    int m0 = mt * 16, n0 = nt * 16;

    stage_a(Xbf + (size_t)m0 * H_SZ, As);
    __syncthreads();

    int half = lane >> 4, l15 = lane & 15;
    v8f acc[4];
#pragma unroll
    for (int g = 0; g < 4; ++g) {
        float bv = bias[g * H_SZ + n0 + l15];
#pragma unroll
        for (int r = 0; r < 8; ++r) acc[g][r] = bv;
    }

    for (int k0 = 0; k0 < H_SZ; k0 += KSTEP) {
        v16bf a = frag_a(As, lane, k0);
#pragma unroll
        for (int g = 0; g < 4; ++g) {
            v16bf b = frag_b(Wt, lane, k0, g * H_SZ + n0);
            acc[g] = __builtin_amdgcn_wmma_f32_16x16x32_bf16(
                false, a, false, b, (short)0, acc[g], false, false);
        }
    }

#pragma unroll
    for (int g = 0; g < 4; ++g) {
#pragma unroll
        for (int r = 0; r < 8; ++r) {
            int m = m0 + r + 8 * half;
            xg[(size_t)m * G4 + g * H_SZ + n0 + l15] = acc[g][r];
        }
    }
}

// ---------------- Phase 2: persistent recurrent scan (one launch per layer) ----------------
// 16 co-resident workgroups; software device barrier between timesteps. Only the
// double-buffered bf16 h crosses workgroups; c/h f32 state is tile-owned (in place).
__global__ __launch_bounds__(256) void k_lstm_scan(
    const float*  __restrict__ xg,       // [T, B, G4]
    const __bf16* __restrict__ Wt,       // [H, G4] (W_hh transposed, bf16)
    __bf16*       __restrict__ hbf,      // [2, B, H] bf16 double buffer (buf0 = h0)
    float*        __restrict__ c_state,  // [B, H] f32 (init c0)
    float*        __restrict__ h_state,  // [B, H] f32
    float*        __restrict__ out_f32,  // [T, B, H] or null (final layer -> d_out)
    __bf16*       __restrict__ out_bf,   // [T, B, H] or null (feeds next layer)
    unsigned int* __restrict__ bar)      // zero-initialized barrier counter
{
    __shared__ __align__(16) __bf16 As[16 * AS_STRIDE];
    const int wave = threadIdx.x >> 5, lane = threadIdx.x & 31;
    const int mt = blockIdx.x >> 2, ntg = blockIdx.x & 3;
    const int nt = ntg * 8 + wave;           // 0..31
    const int m0 = mt * 16, n0 = nt * 16;
    const int half = lane >> 4, l15 = lane & 15;
    const size_t BH = (size_t)B_SZ * H_SZ;

    for (int t = 0; t < T_STEPS; ++t) {
        const __bf16* h_cur  = hbf + (size_t)(t & 1) * BH;
        __bf16*       h_next = hbf + (size_t)((t + 1) & 1) * BH;
        const float*  xg_t   = xg + (size_t)t * B_SZ * G4;

        stage_a(h_cur + (size_t)m0 * H_SZ, As);
        __syncthreads();

        v8f acc[4];
#pragma unroll
        for (int g = 0; g < 4; ++g) {
#pragma unroll
            for (int r = 0; r < 8; ++r) {
                int m = m0 + r + 8 * half;
                acc[g][r] = xg_t[(size_t)m * G4 + g * H_SZ + n0 + l15];
            }
        }

        for (int k0 = 0; k0 < H_SZ; k0 += KSTEP) {
            v16bf a = frag_a(As, lane, k0);
#pragma unroll
            for (int g = 0; g < 4; ++g) {
                v16bf b = frag_b(Wt, lane, k0, g * H_SZ + n0);
                acc[g] = __builtin_amdgcn_wmma_f32_16x16x32_bf16(
                    false, a, false, b, (short)0, acc[g], false, false);
            }
        }

#pragma unroll
        for (int r = 0; r < 8; ++r) {
            int b = m0 + r + 8 * half;       // batch index
            int j = n0 + l15;                // hidden index
            size_t idx = (size_t)b * H_SZ + j;
            float iv = sigmoidf_(acc[0][r]);
            float fv = sigmoidf_(acc[1][r]);
            float gv = tanhf(acc[2][r]);
            float ov = sigmoidf_(acc[3][r]);
            float cn = fv * c_state[idx] + iv * gv;
            float hn = ov * tanhf(cn);
            c_state[idx] = cn;
            h_state[idx] = hn;
            h_next[idx]  = (__bf16)hn;
            if (out_f32) out_f32[(size_t)t * BH + idx] = hn;
            if (out_bf)  out_bf[(size_t)t * BH + idx]  = (__bf16)hn;
        }

        // Prefetch next step's accumulator-init rows (global_prefetch_b8).
        if (t + 1 < T_STEPS) {
            const float* nx = xg + (size_t)(t + 1) * B_SZ * G4 + (size_t)(m0 + l15) * G4 + n0;
            __builtin_prefetch(nx + (half * 2 + 0) * H_SZ, 0, 0);
            __builtin_prefetch(nx + (half * 2 + 1) * H_SZ, 0, 0);
        }

        // Device-scope barrier: release my stores, arrive, spin, acquire.
        __threadfence();                     // release (global_wb at dev scope)
        __syncthreads();
        if (threadIdx.x == 0) {
            atomicAdd(bar, 1u);
            unsigned int target = (unsigned int)(SCAN_BLOCKS * (t + 1));
            while (__hip_atomic_load(bar, __ATOMIC_ACQUIRE, __HIP_MEMORY_SCOPE_AGENT) < target) {
                __builtin_amdgcn_s_sleep(2);
            }
        }
        __syncthreads();
        __threadfence();                     // acquire (global_inv) before reading h_next
    }
}

// ---------------- host ----------------

extern "C" void kernel_launch(void* const* d_in, const int* in_sizes, int n_in,
                              void* d_out, int out_size, void* d_ws, size_t ws_size,
                              hipStream_t stream) {
    (void)in_sizes; (void)n_in; (void)out_size; (void)ws_size;
    const float* x    = (const float*)d_in[0];
    const float* h0   = (const float*)d_in[1];
    const float* c0   = (const float*)d_in[2];
    const float* W_ih = (const float*)d_in[3];
    const float* W_hh = (const float*)d_in[4];
    const float* b_ih = (const float*)d_in[5];
    const float* b_hh = (const float*)d_in[6];
    float* out = (float*)d_out;

    const size_t M = (size_t)T_STEPS * B_SZ;             // 65536
    const size_t BH = (size_t)B_SZ * H_SZ;               // 32768
    const size_t WSZ = (size_t)H_SZ * G4;                // 1M elems per weight matrix

    // workspace carve-up (256B aligned)
    char* ws = (char*)d_ws;
    size_t off = 0;
    auto carve = [&](size_t bytes) -> void* {
        void* p = ws + off;
        off = (off + bytes + 255) & ~(size_t)255;
        return p;
    };
    float*        xg   = (float*)       carve(M * G4 * sizeof(float));      // 512 MB
    __bf16*       xbf  = (__bf16*)      carve(M * H_SZ * sizeof(__bf16));   // 64 MB
    __bf16*       wtih = (__bf16*)      carve(L_SZ * WSZ * sizeof(__bf16)); // 4 MB
    __bf16*       wthh = (__bf16*)      carve(L_SZ * WSZ * sizeof(__bf16)); // 4 MB
    float*        bias = (float*)       carve(L_SZ * G4 * sizeof(float));
    __bf16*       hbf  = (__bf16*)      carve(2 * BH * sizeof(__bf16));     // double buffer
    float*        cst  = (float*)       carve(BH * sizeof(float));
    float*        hst  = (float*)       carve(BH * sizeof(float));
    unsigned int* bar  = (unsigned int*)carve(sizeof(unsigned int));

    // Weight prep: transpose to K-major + bf16, combine biases.
    for (int l = 0; l < L_SZ; ++l) {
        k_transpose_convert<<<(int)(WSZ / 256), 256, 0, stream>>>(
            W_ih + (size_t)l * WSZ, wtih + (size_t)l * WSZ);
        k_transpose_convert<<<(int)(WSZ / 256), 256, 0, stream>>>(
            W_hh + (size_t)l * WSZ, wthh + (size_t)l * WSZ);
        k_bias_sum<<<G4 / 256, 256, 0, stream>>>(b_ih + (size_t)l * G4, b_hh + (size_t)l * G4,
                                                 bias + (size_t)l * G4);
    }

    // Layer 0 input -> bf16
    k_f32_to_bf16<<<(int)(M * H_SZ / 256), 256, 0, stream>>>(x, xbf, (int)(M * H_SZ));

    for (int l = 0; l < L_SZ; ++l) {
        // Phase 1: bulk input projection for all timesteps
        k_gates_pre<<<(int)((M / 16) * 4), 256, 0, stream>>>(
            xbf, wtih + (size_t)l * WSZ, bias + (size_t)l * G4, xg);

        // init recurrent state + barrier
        k_f32_to_bf16<<<(int)(BH / 256), 256, 0, stream>>>(h0 + (size_t)l * BH, hbf, (int)BH);
        hipMemcpyAsync(cst, c0 + (size_t)l * BH, BH * sizeof(float),
                       hipMemcpyDeviceToDevice, stream);
        hipMemcpyAsync(hst, h0 + (size_t)l * BH, BH * sizeof(float),
                       hipMemcpyDeviceToDevice, stream);
        k_zero_u32<<<1, 32, 0, stream>>>(bar);

        // Phase 2: persistent scan over all T steps in one launch
        const bool last = (l == L_SZ - 1);
        k_lstm_scan<<<SCAN_BLOCKS, 256, 0, stream>>>(
            xg, wthh + (size_t)l * WSZ, hbf, cst, hst,
            last ? out : (float*)nullptr,
            last ? (__bf16*)nullptr : xbf,
            bar);

        // h_n, c_n for this layer
        hipMemcpyAsync(out + M * H_SZ + (size_t)l * BH, hst, BH * sizeof(float),
                       hipMemcpyDeviceToDevice, stream);
        hipMemcpyAsync(out + M * H_SZ + (size_t)L_SZ * BH + (size_t)l * BH, cst,
                       BH * sizeof(float), hipMemcpyDeviceToDevice, stream);
    }
}
